// CTAttention_83932250898948
// MI455X (gfx1250) — matrix operand order repaired and checked by reading, not compiled, not run on hardware
//
#include <hip/hip_runtime.h>
#include <hip/hip_bf16.h>

typedef __attribute__((ext_vector_type(16))) _Float16 v16h;
typedef __attribute__((ext_vector_type(8)))  _Float16 v8h;
typedef __attribute__((ext_vector_type(8)))  float    v8f;
typedef __attribute__((ext_vector_type(4)))  float    v4f;

#define B_     8
#define NMAX_  1024
#define C_     256
#define H_     8
#define HD_    32
#define SCALE_ 0.17677669529663687f   // 32^-0.5

static __device__ __forceinline__ v8f wmma_f16(v16h a, v16h b, v8f c) {
  // 8 args: (neg_a, A, neg_b, B, c_mod, C, reuse_a, reuse_b)
  return __builtin_amdgcn_wmma_f32_16x16x32_f16(false, a, false, b, (short)0, c,
                                                false, false);
}

// ---------------------------------------------------------------------------
// Kernel 0: recover per-batch lengths + exclusive row offsets from batch/pos.
// meta[0..7] = len[b], meta[8..15] = row offset of batch b in ragged layout.
// ---------------------------------------------------------------------------
__global__ void ct_len_kernel(const int* __restrict__ batch_id,
                              const int* __restrict__ pos_id,
                              int total, int* __restrict__ meta) {
  if (threadIdx.x < 16) meta[threadIdx.x] = 0;
  __syncthreads();
  for (int i = threadIdx.x; i < total; i += blockDim.x)
    atomicMax(&meta[batch_id[i]], pos_id[i] + 1);
  __syncthreads();
  if (threadIdx.x == 0) {
    int acc = 0;
    for (int b = 0; b < B_; ++b) { meta[8 + b] = acc; acc += meta[b]; }
  }
}

// ---------------------------------------------------------------------------
// Kernel 1: QKV GEMM  [TOTAL,256] x [256,768] + bias, f16 WMMA, f32 accum.
// Scatters into padded per-head tensors:
//   qpad [B*H][NMAX][HD]  (SCALE folded in)
//   ktr  [B*H][NMAX][HD]  (key-major like qpad; QK^T B-operand reads hd contiguously)
//   vpad [B*H][NMAX][HD]
// Block = 4 waves; wave owns 4 N-tiles (64 cols); grid = (TOTAL/16, 768/256).
// ---------------------------------------------------------------------------
__global__ __launch_bounds__(128)
void ct_qkv_kernel(const float* __restrict__ data,
                   const float* __restrict__ qkv_w,
                   const float* __restrict__ qkv_b,
                   const int* __restrict__ batch_id,
                   const int* __restrict__ pos_id,
                   _Float16* __restrict__ qpad,
                   _Float16* __restrict__ ktr,
                   _Float16* __restrict__ vpad) {
  const int lane = threadIdx.x & 31;
  const int wave = threadIdx.x >> 5;
  const int lo = lane & 15, hi = lane >> 4;
  const int row0 = blockIdx.x * 16;
  const int n0 = blockIdx.y * 256 + wave * 64;

  v8f acc[4] = {};
  for (int k0 = 0; k0 < C_; k0 += 32) {
    // A operand: 16x32 tile of data rows, f32 -> f16.
    // Lane holds row (row0+lo); K pattern {hi*8+0..7, 16+hi*8+0..7}.
    const float* ar = data + (size_t)(row0 + lo) * C_ + k0 + hi * 8;
    v4f a0 = *(const v4f*)(ar);
    v4f a1 = *(const v4f*)(ar + 4);
    v4f a2 = *(const v4f*)(ar + 16);
    v4f a3 = *(const v4f*)(ar + 20);
    v16h a;
#pragma unroll
    for (int j = 0; j < 4; ++j) {
      a[j]      = (_Float16)a0[j];
      a[4 + j]  = (_Float16)a1[j];
      a[8 + j]  = (_Float16)a2[j];
      a[12 + j] = (_Float16)a3[j];
    }
    // B operand: lane = N col (n0+t*16+lo), elements = K {hi*8+0..7, 16+hi*8+0..7}.
    const float* br = qkv_w + (size_t)(k0 + hi * 8) * (3 * C_) + n0 + lo;
#pragma unroll
    for (int t = 0; t < 4; ++t) {
      v16h bb;
#pragma unroll
      for (int j = 0; j < 8; ++j) {
        bb[j]     = (_Float16)br[(size_t)j * (3 * C_) + t * 16];
        bb[8 + j] = (_Float16)br[(size_t)(16 + j) * (3 * C_) + t * 16];
      }
      acc[t] = wmma_f16(a, bb, acc[t]);
    }
  }

  // Epilogue: D layout lane = N (col), VGPR v = row (hi*8+v). Route q/k/v.
#pragma unroll
  for (int t = 0; t < 4; ++t) {
    const int ncol = n0 + t * 16 + lo;
    const int which = ncol >> 8;          // 0=q 1=k 2=v (uniform per block)
    const int h  = (ncol & 255) >> 5;
    const int hd = ncol & 31;
    const float bias = qkv_b[ncol];
#pragma unroll
    for (int v = 0; v < 8; ++v) {
      const int row = row0 + hi * 8 + v;
      const int b   = batch_id[row];
      const int pos = pos_id[row];
      const int bh  = b * H_ + h;
      const float val = acc[t][v] + bias;
      if (which == 0)
        qpad[((size_t)bh * NMAX_ + pos) * HD_ + hd] = (_Float16)(val * SCALE_);
      else if (which == 1)
        ktr[((size_t)bh * NMAX_ + pos) * HD_ + hd] = (_Float16)val;
      else
        vpad[((size_t)bh * NMAX_ + pos) * HD_ + hd] = (_Float16)val;
    }
  }
}

// ---------------------------------------------------------------------------
// Kernel 2: flash attention. One wave per (b,h, 16-query tile).
// Streams keys in tiles of 32 (all lengths divisible by 32 -> no partial
// tiles, EXEC stays all-ones around every WMMA). Padded keys are skipped:
// exp(-1e4) == 0 in fp32, so this matches the masked reference exactly.
// ---------------------------------------------------------------------------
__global__ __launch_bounds__(32)
void ct_attn_kernel(const _Float16* __restrict__ qpad,
                    const _Float16* __restrict__ ktr,
                    const _Float16* __restrict__ vpad,
                    const int* __restrict__ meta,
                    _Float16* __restrict__ aout) {
  const int bh = blockIdx.x;
  const int b = bh >> 3, h = bh & 7;
  const int len = meta[b];
  const int q0 = blockIdx.y * 16;
  if (q0 >= len) return;                  // uniform over the wave
  const int offs = meta[8 + b];

  const int lane = threadIdx.x;
  const int lo = lane & 15, hi = lane >> 4;

  __shared__ _Float16 pt[16 * 32];        // P tile, D-layout -> A-layout xpose

  // Q A-operand, loaded once (SCALE already folded in).
  const _Float16* qr = qpad + ((size_t)bh * NMAX_ + q0 + lo) * HD_ + hi * 8;
  v16h aq;
  {
    v8h qa = *(const v8h*)(qr);
    v8h qb = *(const v8h*)(qr + 16);
#pragma unroll
    for (int j = 0; j < 8; ++j) { aq[j] = qa[j]; aq[8 + j] = qb[j]; }
  }

  v8f o0 = {}, o1 = {};
  const v8f vzero = {};
  float m[8], l[8];
#pragma unroll
  for (int v = 0; v < 8; ++v) { m[v] = -3.0e38f; l[v] = 0.f; }

  const _Float16* krow = ktr + ((size_t)bh * NMAX_ + lo) * HD_ + hi * 8; // lane = key
  const _Float16* vbase = vpad + (size_t)bh * NMAX_ * HD_;

  for (int kt = 0; kt < len; kt += 32) {
    if (kt + 32 < len) {                  // prefetch next K/V tiles
      __builtin_prefetch(krow + (size_t)(kt + 32) * HD_, 0, 0);
      __builtin_prefetch(vbase + (size_t)(kt + 32 + lane) * HD_, 0, 0);
    }
    // S = Q K^T : B-operand lane = key, elements = hd {hi*8+0..7, 16+hi*8+0..7}.
    v16h bk0, bk1;
    {
      const _Float16* k0p = krow + (size_t)kt * HD_;
      const _Float16* k1p = krow + (size_t)(kt + 16) * HD_;
      v8h ka = *(const v8h*)(k0p), kb = *(const v8h*)(k0p + 16);
      v8h kc = *(const v8h*)(k1p), kd = *(const v8h*)(k1p + 16);
#pragma unroll
      for (int j = 0; j < 8; ++j) {
        bk0[j] = ka[j]; bk0[8 + j] = kb[j];
        bk1[j] = kc[j]; bk1[8 + j] = kd[j];
      }
    }
    v8f s0 = wmma_f16(aq, bk0, vzero);
    v8f s1 = wmma_f16(aq, bk1, vzero);

    // Online softmax. Row (hi*8+v) lives in s0[v]/s1[v] across 16 lanes.
    float p0[8], p1[8];
#pragma unroll
    for (int v = 0; v < 8; ++v) {
      float mx = fmaxf(s0[v], s1[v]);
#pragma unroll
      for (int d = 1; d < 16; d <<= 1) mx = fmaxf(mx, __shfl_xor(mx, d, 32));
      const float mn = fmaxf(m[v], mx);
      const float sc = __expf(m[v] - mn);
      const float e0 = __expf(s0[v] - mn);
      const float e1 = __expf(s1[v] - mn);
      float sum = e0 + e1;
#pragma unroll
      for (int d = 1; d < 16; d <<= 1) sum += __shfl_xor(sum, d, 32);
      l[v] = l[v] * sc + sum;
      m[v] = mn;
      o0[v] *= sc;
      o1[v] *= sc;
      p0[v] = e0;
      p1[v] = e1;
    }

    // Transpose P (D-layout) -> A-layout through LDS.
#pragma unroll
    for (int v = 0; v < 8; ++v) {
      pt[(hi * 8 + v) * 32 + lo]      = (_Float16)p0[v];
      pt[(hi * 8 + v) * 32 + 16 + lo] = (_Float16)p1[v];
    }
    __syncthreads();
    v16h ap;
    {
      const _Float16* pr = pt + lo * 32 + hi * 8;
      v8h pa = *(const v8h*)(pr);
      v8h pb = *(const v8h*)(pr + 16);
#pragma unroll
      for (int j = 0; j < 8; ++j) { ap[j] = pa[j]; ap[8 + j] = pb[j]; }
    }

    // O += P V : B-operand lane = hd, elements = keys {hi*8+0..7, 16+hi*8+0..7}.
    const _Float16* vr = vbase + (size_t)(kt + hi * 8) * HD_ + lo;
    v16h bv0, bv1;
#pragma unroll
    for (int j = 0; j < 8; ++j) {
      bv0[j] = vr[j * HD_];      bv0[8 + j] = vr[(16 + j) * HD_];
      bv1[j] = vr[j * HD_ + 16]; bv1[8 + j] = vr[(16 + j) * HD_ + 16];
    }
    o0 = wmma_f16(ap, bv0, o0);
    o1 = wmma_f16(ap, bv1, o1);
  }

  // Normalize and write unpadded: row = offs + q, col = h*32 + hd.
#pragma unroll
  for (int v = 0; v < 8; ++v) {
    const float inv = 1.0f / l[v];
    const int row = offs + q0 + hi * 8 + v;
    _Float16* orow = aout + (size_t)row * C_ + h * HD_;
    orow[lo]      = (_Float16)(o0[v] * inv);
    orow[16 + lo] = (_Float16)(o1[v] * inv);
  }
}

// ---------------------------------------------------------------------------
// Kernel 3: projection GEMM [TOTAL,256] x [256,256] + bias -> f32 out.
// ---------------------------------------------------------------------------
__global__ __launch_bounds__(128)
void ct_proj_kernel(const _Float16* __restrict__ aout,
                    const float* __restrict__ proj_w,
                    const float* __restrict__ proj_b,
                    float* __restrict__ out) {
  const int lane = threadIdx.x & 31;
  const int wave = threadIdx.x >> 5;
  const int lo = lane & 15, hi = lane >> 4;
  const int row0 = blockIdx.x * 16;
  const int n0 = wave * 64;

  v8f acc[4] = {};
  for (int k0 = 0; k0 < C_; k0 += 32) {
    const _Float16* ar = aout + (size_t)(row0 + lo) * C_ + k0 + hi * 8;
    v16h a;
    {
      v8h a0 = *(const v8h*)(ar);
      v8h a1 = *(const v8h*)(ar + 16);
#pragma unroll
      for (int j = 0; j < 8; ++j) { a[j] = a0[j]; a[8 + j] = a1[j]; }
    }
    const float* br = proj_w + (size_t)(k0 + hi * 8) * C_ + n0 + lo;
#pragma unroll
    for (int t = 0; t < 4; ++t) {
      v16h bb;
#pragma unroll
      for (int j = 0; j < 8; ++j) {
        bb[j]     = (_Float16)br[(size_t)j * C_ + t * 16];
        bb[8 + j] = (_Float16)br[(size_t)(16 + j) * C_ + t * 16];
      }
      acc[t] = wmma_f16(a, bb, acc[t]);
    }
  }
#pragma unroll
  for (int t = 0; t < 4; ++t) {
    const int ncol = n0 + t * 16 + lo;
    const float bias = proj_b[ncol];
#pragma unroll
    for (int v = 0; v < 8; ++v)
      out[(size_t)(row0 + hi * 8 + v) * C_ + ncol] = acc[t][v] + bias;
  }
}

// ---------------------------------------------------------------------------
extern "C" void kernel_launch(void* const* d_in, const int* in_sizes, int n_in,
                              void* d_out, int out_size, void* d_ws,
                              size_t ws_size, hipStream_t stream) {
  const float* data   = (const float*)d_in[0];
  const float* qkv_w  = (const float*)d_in[1];
  const float* qkv_b  = (const float*)d_in[2];
  const float* proj_w = (const float*)d_in[3];
  const float* proj_b = (const float*)d_in[4];
  // d_in[5] = ct_mask: unused — padded keys contribute exactly 0 in fp32,
  // equivalent masking is derived from lengths (batch_id/pos_id).
  const int* batch_id = (const int*)d_in[6];
  const int* pos_id   = (const int*)d_in[7];
  float* out = (float*)d_out;

  const int total = in_sizes[6];                        // TOTAL (6656)
  char* ws = (char*)d_ws;
  int* meta = (int*)ws;                                 // 16 ints
  const size_t qkv_elems = (size_t)B_ * H_ * NMAX_ * HD_;  // 2M f16 each
  _Float16* qpad = (_Float16*)(ws + 256);
  _Float16* ktr  = qpad + qkv_elems;
  _Float16* vpad = ktr + qkv_elems;
  _Float16* aout = vpad + qkv_elems;                    // [TOTAL, C] f16

  ct_len_kernel<<<1, 256, 0, stream>>>(batch_id, pos_id, total, meta);
  ct_qkv_kernel<<<dim3(total / 16, 3), 128, 0, stream>>>(
      data, qkv_w, qkv_b, batch_id, pos_id, qpad, ktr, vpad);
  ct_attn_kernel<<<dim3(B_ * H_, NMAX_ / 16), 32, 0, stream>>>(
      qpad, ktr, vpad, meta, aout);
  ct_proj_kernel<<<dim3(total / 16), 128, 0, stream>>>(aout, proj_w, proj_b,
                                                       out);
}
